// Attention_57483842289792
// MI455X (gfx1250) — compile-verified, matrix operand order checked by reading, not run on hardware
//
#include <hip/hip_runtime.h>

// ---------------- problem constants ----------------
#define BB   8
#define SS   1024
#define DD   768
#define HH   12
#define DHH  64
#define MMR  (BB * SS)        // 8192 rows
#define NQKV (3 * DD)         // 2304

typedef __attribute__((ext_vector_type(16))) __bf16 v16bf;
typedef __attribute__((ext_vector_type(8)))  __bf16 v8bf;
typedef __attribute__((ext_vector_type(8)))  float  v8f;
typedef __attribute__((ext_vector_type(4)))  unsigned int v4u;
typedef __attribute__((ext_vector_type(8)))  unsigned int v8u;

// Load a 16x32 bf16 A/B fragment row for one lane.
// CDNA5 layout: lanes 0-15 hold K = [0..7, 16..23], lanes 16-31 hold K = [8..15, 24..31].
__device__ __forceinline__ v16bf load_frag_row(const __bf16* rowbase, int half) {
    const v8bf lo = *(const v8bf*)(rowbase + half * 8);
    const v8bf hi = *(const v8bf*)(rowbase + 16 + half * 8);
    return __builtin_shufflevector(lo, hi, 0, 1, 2, 3, 4, 5, 6, 7,
                                           8, 9, 10, 11, 12, 13, 14, 15);
}

// ---------------- Tensor Data Mover: 2D tile Global -> LDS ----------------
// D# group0 (4 SGPRs) + group1 (8 SGPRs) per CDNA5 ISA.
// 2-operand form (groups 2/3 NULL) = 2D tensors, all we need.
// pad_interval code: 2^(code+1) DWORDs; pad_amount code: code+1 DWORDs.
__device__ __forceinline__ void tdm_load_2d(unsigned lds_off, const void* gaddr,
                                            unsigned tile_d0, unsigned tile_d1,
                                            unsigned stride0,
                                            unsigned pad_interval_code,
                                            unsigned pad_amount_code) {
    unsigned long long ga = (unsigned long long)gaddr;
    v4u g0;
    g0[0] = 1u;                                                // count=1, user mode
    g0[1] = lds_off;                                           // lds_addr (bytes)
    g0[2] = (unsigned)(ga & 0xffffffffu);                      // global_addr[31:0]
    g0[3] = (unsigned)((ga >> 32) & 0x01ffffffu) | (2u << 30); // addr[56:32] | type=2
    v8u g1;
    g1[0] = (1u << 16)                        // data_size = 1 -> 2 bytes
          | (1u << 20)                        // pad_enable
          | (pad_interval_code << 22)
          | (pad_amount_code << 25);
    g1[1] = (tile_d0 & 0xffffu) << 16;        // tensor_dim0[15:0] (== tile_d0)
    g1[2] = (tile_d0 >> 16) | ((tile_d1 & 0xffffu) << 16);   // dim0 hi | tensor_dim1 lo
    g1[3] = (tile_d1 >> 16) | (tile_d0 << 16);               // dim1 hi | tile_dim0
    g1[4] = tile_d1 & 0xffffu;                // tile_dim1 (tile_dim2 = 0 -> 2D)
    g1[5] = stride0;                          // tensor_dim0_stride[31:0] (elements)
    g1[6] = 0u;
    g1[7] = 0u;
    asm volatile("tensor_load_to_lds %0, %1" :: "s"(g0), "s"(g1) : "memory");
}

// ---------------- pack / cast kernels ----------------
__global__ void cast_bf16_kernel(const float* __restrict__ s, __bf16* __restrict__ d, int n) {
    int i = blockIdx.x * blockDim.x + threadIdx.x;
    if (i < n) d[i] = (__bf16)s[i];
}

// Wqkv_t rows: n = which*768 + h*64 + e, cols k = d.  Wt[n][k] = W_which[h][k][e]
__global__ void pack_wqkv_kernel(const float* __restrict__ Wq, const float* __restrict__ Wk,
                                 const float* __restrict__ Wv, __bf16* __restrict__ dst) {
    int i = blockIdx.x * blockDim.x + threadIdx.x;
    if (i >= NQKV * DD) return;
    int n = i / DD, d = i - (i / DD) * DD;
    int which = n / DD;
    int rem = n - which * DD;
    int h = rem >> 6, e = rem & 63;
    const float* W = (which == 0) ? Wq : (which == 1) ? Wk : Wv;
    dst[i] = (__bf16)W[((size_t)h * DD + d) * DHH + e];
}

// Wo_t rows: n = d (output dim), cols k = h*64+e.  Wt[n][k] = W_O[h][e][n]
__global__ void pack_wo_kernel(const float* __restrict__ Wo, __bf16* __restrict__ dst) {
    int i = blockIdx.x * blockDim.x + threadIdx.x;
    if (i >= DD * DD) return;
    int n = i / DD, k = i - (i / DD) * DD;
    int h = k >> 6, e = k & 63;
    dst[i] = (__bf16)Wo[((size_t)h * DHH + e) * DD + n];
}

// ---------------- WMMA GEMM: C[M,N] = A[M,K] * Bt[N,K]^T ----------------
// Double-buffered TDM pipeline: wave 0 issues the next tile pair, then
// s_wait_tensorcnt(2) retires the current pair (tensor ops complete in order
// per wave) while the new pair stays in flight under the WMMA compute.
// EPI=0: QKV epilogue (scatter to q/k/v [B,H,S,DH] bf16, +bias, q scaled 1/8)
// EPI=1: fp32 epilogue to out[m*N+n] + bias0[n]
template <int EPI>
__global__ __launch_bounds__(256) void gemm_wmma_kernel(
    const __bf16* __restrict__ A, const __bf16* __restrict__ Bt, int N, int K,
    const float* __restrict__ bias0, const float* __restrict__ bias1,
    const float* __restrict__ bias2,
    __bf16* __restrict__ qo, __bf16* __restrict__ ko, __bf16* __restrict__ vo,
    float* __restrict__ fo) {
    constexpr int BK  = 32;
    constexpr int LDT = 40;  // bf16 elems per LDS row (80 bytes; TDM pad codes 3/3)
    __shared__ alignas(16) __bf16 sA[2][128 * LDT];
    __shared__ alignas(16) __bf16 sB[2][128 * LDT];

    const int tid  = threadIdx.x;
    const int lane = tid & 31;
    const int wave = tid >> 5;   // 0..7
    const int wm   = wave >> 2;  // 0..1
    const int wn   = wave & 3;   // 0..3
    const int half = lane >> 4;
    const int l16  = lane & 15;

    const int m0 = blockIdx.y * 128;
    const int n0 = blockIdx.x * 128;

    unsigned sA_off[2] = {(unsigned)(size_t)(void*)sA[0], (unsigned)(size_t)(void*)sA[1]};
    unsigned sB_off[2] = {(unsigned)(size_t)(void*)sB[0], (unsigned)(size_t)(void*)sB[1]};

    v8f acc[4][2];
#pragma unroll
    for (int i = 0; i < 4; ++i)
#pragma unroll
        for (int j = 0; j < 2; ++j) acc[i][j] = {};

    const int nk = K / BK;
    if (wave == 0) {  // prologue: stage tile 0 into buffer 0
        tdm_load_2d(sA_off[0], &A[(size_t)m0 * K], BK, 128, (unsigned)K, 3, 3);
        tdm_load_2d(sB_off[0], &Bt[(size_t)n0 * K], BK, 128, (unsigned)K, 3, 3);
    }
    for (int kt = 0; kt < nk; ++kt) {
        const int buf = kt & 1;
        __syncthreads();  // all waves done reading buffer (1-buf) -> safe to refill
        if (wave == 0) {
            if (kt + 1 < nk) {
                const int k1 = (kt + 1) * BK;
                tdm_load_2d(sA_off[1 - buf], &A[(size_t)m0 * K + k1], BK, 128, (unsigned)K, 3, 3);
                tdm_load_2d(sB_off[1 - buf], &Bt[(size_t)n0 * K + k1], BK, 128, (unsigned)K, 3, 3);
                __builtin_amdgcn_s_wait_tensorcnt(2);  // current pair landed; next in flight
            } else {
                __builtin_amdgcn_s_wait_tensorcnt(0);
            }
        }
        __syncthreads();  // publish buffer buf

        v16bf af[4], bfr[2];
#pragma unroll
        for (int mi = 0; mi < 4; ++mi)
            af[mi] = load_frag_row(&sA[buf][(wm * 64 + mi * 16 + l16) * LDT], half);
#pragma unroll
        for (int ni = 0; ni < 2; ++ni)
            bfr[ni] = load_frag_row(&sB[buf][(wn * 32 + ni * 16 + l16) * LDT], half);
#pragma unroll
        for (int mi = 0; mi < 4; ++mi)
#pragma unroll
            for (int ni = 0; ni < 2; ++ni)
                acc[mi][ni] = __builtin_amdgcn_wmma_f32_16x16x32_bf16(
                    false, af[mi], false, bfr[ni], (short)0, acc[mi][ni], false, false);
    }

    // epilogue: C layout = VGPR j -> row (j + 8*half), lane&15 -> col
#pragma unroll
    for (int mi = 0; mi < 4; ++mi) {
#pragma unroll
        for (int ni = 0; ni < 2; ++ni) {
#pragma unroll
            for (int j = 0; j < 8; ++j) {
                int m    = m0 + wm * 64 + mi * 16 + j + half * 8;
                int n    = n0 + wn * 32 + ni * 16 + l16;
                float v  = acc[mi][ni][j];
                if (EPI == 0) {
                    int which = n / DD;
                    int rem   = n - which * DD;
                    int h = rem >> 6, e = rem & 63;
                    int b = m >> 10, s = m & 1023;
                    const float* bias = (which == 0) ? bias0 : (which == 1) ? bias1 : bias2;
                    v += bias[rem];
                    if (which == 0) v *= 0.125f;  // fold 1/sqrt(DH) into Q
                    __bf16* dst = (which == 0) ? qo : (which == 1) ? ko : vo;
                    dst[(((size_t)b * HH + h) * SS + s) * DHH + e] = (__bf16)v;
                } else {
                    fo[(size_t)m * N + n] = v + bias0[n];
                }
            }
        }
    }
}

// ---------------- flash attention: per (b, h, 64-row q tile) ----------------
__global__ __launch_bounds__(128) void flash_attn_kernel(
    const __bf16* __restrict__ Q, const __bf16* __restrict__ Kg,
    const __bf16* __restrict__ Vg, __bf16* __restrict__ Z) {
    constexpr int LDT = 72;  // 144-byte rows (TDM pad codes 4/3)
    __shared__ alignas(16) __bf16 sK[2][64 * LDT];   // K tiles, TDM double-buffered
    __shared__ alignas(16) __bf16 sV[64 * LDT];      // V tile transposed [e][k]
    __shared__ alignas(16) __bf16 sP[4][16 * LDT];   // per-wave P transpose buffer

    const int tid  = threadIdx.x;
    const int lane = tid & 31;
    const int wave = tid >> 5;  // 0..3 -> q rows [q0+wave*16, +16)
    const int half = lane >> 4;
    const int l16  = lane & 15;

    const int qt = blockIdx.x;   // 0..15
    const int h  = blockIdx.y;
    const int b  = blockIdx.z;
    const int q0 = qt * 64;

    const size_t headoff = ((size_t)b * HH + h) * SS * DHH;
    const __bf16* Qh = Q + headoff;
    const __bf16* Kh = Kg + headoff;
    const __bf16* Vh = Vg + headoff;

    unsigned sK_off[2] = {(unsigned)(size_t)(void*)sK[0], (unsigned)(size_t)(void*)sK[1]};

    // hold Q (pre-scaled by 1/8) in registers for the whole loop
    v16bf qf[2];
    {
        const __bf16* rb = &Qh[(size_t)(q0 + wave * 16 + l16) * DHH];
        qf[0] = load_frag_row(rb, half);       // e in [0,32)
        qf[1] = load_frag_row(rb + 32, half);  // e in [32,64)
    }

    v8f oacc[4];
#pragma unroll
    for (int i = 0; i < 4; ++i) oacc[i] = {};
    float mrun[8], lrun[8];
#pragma unroll
    for (int j = 0; j < 8; ++j) { mrun[j] = -3.0e38f; lrun[j] = 0.0f; }

    if (wave == 0)  // prologue: K tile 0 -> buffer 0
        tdm_load_2d(sK_off[0], Kh, DHH, 64, DHH, 4, 3);

    for (int ktile = 0; ktile <= qt; ++ktile) {
        const int k0  = ktile * 64;
        const int buf = ktile & 1;
        __syncthreads();  // previous compute done -> safe to refill K[1-buf] and sV
        if (wave == 0) {
            if (ktile < qt) {
                tdm_load_2d(sK_off[1 - buf], &Kh[(size_t)(k0 + 64) * DHH], DHH, 64, DHH, 4, 3);
                __builtin_amdgcn_s_wait_tensorcnt(1);  // current K landed; next in flight
            } else {
                __builtin_amdgcn_s_wait_tensorcnt(0);
            }
        }
        // V tile staged transposed by all threads (overlaps wave 0's tensor wait)
#pragma unroll
        for (int it = 0; it < 4; ++it) {
            int g = tid + it * 128;
            int r = g >> 3;
            int c = (g & 7) * 8;
            uint4 vv = *(const uint4*)(&Vh[(size_t)(k0 + r) * DHH + c]);
            const __bf16* vp = (const __bf16*)&vv;
#pragma unroll
            for (int x = 0; x < 8; ++x) sV[(c + x) * LDT + r] = vp[x];
        }
        if (ktile < qt)  // warm GL2 for next V tile (global_prefetch_b8)
            __builtin_prefetch(&Vh[(size_t)(k0 + 64) * DHH + tid * 64], 0, 1);
        __syncthreads();

        // S = Q * K^T : batch all 8 K fragments, then run the WMMA chain
        v16bf kf[4][2];
#pragma unroll
        for (int nt = 0; nt < 4; ++nt)
#pragma unroll
            for (int ks = 0; ks < 2; ++ks)
                kf[nt][ks] = load_frag_row(&sK[buf][(nt * 16 + l16) * LDT + ks * 32], half);
        v8f sacc[4];
#pragma unroll
        for (int i = 0; i < 4; ++i) sacc[i] = {};
#pragma unroll
        for (int nt = 0; nt < 4; ++nt)
#pragma unroll
            for (int ks = 0; ks < 2; ++ks)
                sacc[nt] = __builtin_amdgcn_wmma_f32_16x16x32_bf16(
                    false, qf[ks], false, kf[nt][ks], (short)0, sacc[nt], false, false);

        // causal mask on the diagonal tile
        if (ktile == qt) {
#pragma unroll
            for (int nt = 0; nt < 4; ++nt)
#pragma unroll
                for (int j = 0; j < 8; ++j) {
                    int mrow = wave * 16 + j + half * 8;
                    int ncol = nt * 16 + l16;
                    if (ncol > mrow) sacc[nt][j] = -100000.0f;
                }
        }

        // online softmax (row stats replicated across each 16-lane N group)
        float pexp[4][8];
#pragma unroll
        for (int j = 0; j < 8; ++j) {
            float tm = fmaxf(fmaxf(sacc[0][j], sacc[1][j]), fmaxf(sacc[2][j], sacc[3][j]));
#pragma unroll
            for (int o = 1; o < 16; o <<= 1) tm = fmaxf(tm, __shfl_xor(tm, o, 32));
            float newm = fmaxf(mrun[j], tm);
            float corr = __expf(mrun[j] - newm);
            float rs = 0.0f;
#pragma unroll
            for (int nt = 0; nt < 4; ++nt) {
                float p = __expf(sacc[nt][j] - newm);
                pexp[nt][j] = p;
                rs += p;
            }
#pragma unroll
            for (int o = 1; o < 16; o <<= 1) rs += __shfl_xor(rs, o, 32);
            lrun[j] = lrun[j] * corr + rs;
            mrun[j] = newm;
#pragma unroll
            for (int i = 0; i < 4; ++i) oacc[i][j] *= corr;
        }

        // transpose P (C layout) -> A layout through per-wave LDS buffer
#pragma unroll
        for (int nt = 0; nt < 4; ++nt)
#pragma unroll
            for (int j = 0; j < 8; ++j)
                sP[wave][(j + half * 8) * LDT + nt * 16 + l16] = (__bf16)pexp[nt][j];
        // (same-wave DS ops are in order; no cross-wave sharing of sP)

        // O += P * V : batch fragments per k-step, then WMMA
#pragma unroll
        for (int ks = 0; ks < 2; ++ks) {
            v16bf pf = load_frag_row(&sP[wave][l16 * LDT + ks * 32], half);
            v16bf vf[4];
#pragma unroll
            for (int et = 0; et < 4; ++et)
                vf[et] = load_frag_row(&sV[(et * 16 + l16) * LDT + ks * 32], half);
#pragma unroll
            for (int et = 0; et < 4; ++et)
                oacc[et] = __builtin_amdgcn_wmma_f32_16x16x32_bf16(
                    false, pf, false, vf[et], (short)0, oacc[et], false, false);
        }
    }

    // finalize: O /= l ; write Z in [B,S,H,DH] (== [8192][768] for the O-proj GEMM)
#pragma unroll
    for (int et = 0; et < 4; ++et)
#pragma unroll
        for (int j = 0; j < 8; ++j) {
            float v = oacc[et][j] / lrun[j];
            int srow = q0 + wave * 16 + j + half * 8;
            int e = et * 16 + l16;
            Z[(((size_t)b * SS + srow) * HH + h) * DHH + e] = (__bf16)v;
        }
}

// ---------------- launch ----------------
extern "C" void kernel_launch(void* const* d_in, const int* in_sizes, int n_in,
                              void* d_out, int out_size, void* d_ws, size_t ws_size,
                              hipStream_t stream) {
    const float* x  = (const float*)d_in[0];
    const float* WQ = (const float*)d_in[1];
    const float* bQ = (const float*)d_in[2];
    const float* WK = (const float*)d_in[3];
    const float* bK = (const float*)d_in[4];
    const float* WV = (const float*)d_in[5];
    const float* bV = (const float*)d_in[6];
    const float* WO = (const float*)d_in[7];
    const float* bO = (const float*)d_in[8];
    float* out = (float*)d_out;

    char* ws = (char*)d_ws;
    const size_t SZ_X  = (size_t)MMR * DD * 2;   // 12,582,912
    const size_t SZ_WQ = (size_t)NQKV * DD * 2;  //  3,538,944
    const size_t SZ_WO = (size_t)DD * DD * 2;    //  1,179,648
    __bf16* x_bf  = (__bf16*)(ws);
    __bf16* wqkv  = (__bf16*)(ws + SZ_X);
    __bf16* wo    = (__bf16*)(ws + SZ_X + SZ_WQ);
    __bf16* q_bf  = (__bf16*)(ws + SZ_X + SZ_WQ + SZ_WO);
    __bf16* k_bf  = (__bf16*)(ws + 2 * SZ_X + SZ_WQ + SZ_WO);
    __bf16* v_bf  = (__bf16*)(ws + 3 * SZ_X + SZ_WQ + SZ_WO);
    __bf16* z_bf  = (__bf16*)(ws + 4 * SZ_X + SZ_WQ + SZ_WO);

    cast_bf16_kernel<<<(MMR * DD + 255) / 256, 256, 0, stream>>>(x, x_bf, MMR * DD);
    pack_wqkv_kernel<<<(NQKV * DD + 255) / 256, 256, 0, stream>>>(WQ, WK, WV, wqkv);
    pack_wo_kernel<<<(DD * DD + 255) / 256, 256, 0, stream>>>(WO, wo);

    // QKV projection: [8192 x 2304 x 768]
    gemm_wmma_kernel<0><<<dim3(NQKV / 128, MMR / 128), 256, 0, stream>>>(
        x_bf, wqkv, NQKV, DD, bQ, bK, bV, q_bf, k_bf, v_bf, nullptr);

    // causal flash attention
    flash_attn_kernel<<<dim3(SS / 64, HH, BB), 128, 0, stream>>>(q_bf, k_bf, v_bf, z_bf);

    // output projection: [8192 x 768 x 768] -> fp32 + b_O
    gemm_wmma_kernel<1><<<dim3(DD / 128, MMR / 128), 256, 0, stream>>>(
        z_bf, wo, DD, DD, bO, nullptr, nullptr, nullptr, nullptr, nullptr, out);
}
